// Upsample_71356586655954
// MI455X (gfx1250) — compile-verified, matrix operand order checked by reading, not run on hardware
//
#include <hip/hip_runtime.h>
#include <hip/hip_bf16.h>
#include <stddef.h>

// ---------------------------------------------------------------------------
// Reference: x[16,256,64,64] -> nearest-up2x -> conv3x3(ternary w, pad1) + bias
// Parity folding: out(2i+a, 2j+b) is a 2x2 conv on the ORIGINAL 64x64 grid.
// GEMM per parity: M=256(co), N=16*64*64=65536(pos), K=256*4=1024.
// ---------------------------------------------------------------------------
#define C_IN    256
#define C_OUT   256
#define NW      589824          // 256*256*9 raw weight elements
#define KF      1024            // folded K = 256*4  (k = ci*4 + dy*2 + dx)
#define OSPAT   4096            // 64*64 original spatial
#define NPOSF   65536           // 16*OSPAT positions per parity

typedef __attribute__((ext_vector_type(16))) __bf16 v16bf;
typedef __attribute__((ext_vector_type(8)))  __bf16 v8bf;
typedef __attribute__((ext_vector_type(8)))  float  v8f;

__device__ __forceinline__ unsigned f2bf(float f) {
    unsigned u = __float_as_uint(f);
    return (u + 0x7FFFu + ((u >> 16) & 1u)) >> 16;   // round-nearest-even
}

// ---------------------------------------------------------------------------
// Kernel 0: sum |w| -> d_acc (pre-zeroed via hipMemsetAsync)
// ---------------------------------------------------------------------------
__global__ void absum_kernel(const float* __restrict__ w, float* __restrict__ acc, int n) {
    __shared__ float s[256];
    float sum = 0.0f;
    for (int i = blockIdx.x * blockDim.x + threadIdx.x; i < n; i += gridDim.x * blockDim.x)
        sum += fabsf(w[i]);
    s[threadIdx.x] = sum;
    __syncthreads();
    for (int o = 128; o > 0; o >>= 1) {
        if ((int)threadIdx.x < o) s[threadIdx.x] += s[threadIdx.x + o];
        __syncthreads();
    }
    if (threadIdx.x == 0) atomicAdd(acc, s[0]);
}

// ---------------------------------------------------------------------------
// Kernel 1: ternarize + parity-fold weights to bf16.
//   qw[p][co][k],  p = a*2+b,  k = ci*4 + dy*2 + dx,  total 4*256*1024 elems.
// ---------------------------------------------------------------------------
__global__ void fold_ternarize_kernel(const float* __restrict__ w,
                                      const float* __restrict__ acc,
                                      unsigned short* __restrict__ qw) {
    const int idx = blockIdx.x * 256 + threadIdx.x;      // 0 .. 4*256*1024-1
    const int k   = idx & (KF - 1);
    const int co  = (idx >> 10) & 255;
    const int p   = idx >> 18;
    const int a   = p >> 1, b = p & 1;
    const int ci  = k >> 2, dy = (k >> 1) & 1, dx = k & 1;

    const float alpha = (*acc) * (1.0f / (float)NW);
    const float thr   = 0.001f * alpha;

    int kh0, khn;
    if (a == 0) { kh0 = dy ? 1 : 0; khn = dy ? 2 : 1; }
    else        { kh0 = dy ? 2 : 0; khn = dy ? 1 : 2; }
    int kw0, kwn;
    if (b == 0) { kw0 = dx ? 1 : 0; kwn = dx ? 2 : 1; }
    else        { kw0 = dx ? 2 : 0; kwn = dx ? 1 : 2; }

    const float* wb = w + ((size_t)co * C_IN + ci) * 9;
    float s = 0.0f;
    for (int r = 0; r < khn; ++r)
        for (int c = 0; c < kwn; ++c) {
            float v = wb[(kh0 + r) * 3 + (kw0 + c)];
            s += (v > thr) ? alpha : ((v < -thr) ? -alpha : 0.0f);
        }
    qw[idx] = (unsigned short)f2bf(s);
}

// ---------------------------------------------------------------------------
// Kernel 2: folded 2x2 conv as implicit GEMM via v_wmma_f32_16x16x32_bf16.
//   grid = (16 = parity*coTile fastest, 256 posTiles); block = 256 thr (8 waves).
//   Block tile: 64 co x 256 pos. Wave tile: 32 co x 64 pos -> 8 WMMA / K-step.
//   A tile staged via GLOBAL_LOAD_ASYNC_TO_LDS_B128 (ASYNCcnt path).
// ---------------------------------------------------------------------------
__global__ __launch_bounds__(256) void conv_wmma_kernel(
        const float* __restrict__ x,            // [16,256,64,64]
        const unsigned short* __restrict__ qw,  // bf16 [4][256][1024]
        const float* __restrict__ bias,         // [256]
        float* __restrict__ out)                // [16,256,128,128]
{
    __shared__ __align__(16) unsigned short lA[64][40];    // [co][k], 80B rows
    __shared__ __align__(16) unsigned short lB[256][40];   // [pos][k]

    const int t    = threadIdx.x;
    const int lane = t & 31;
    const int wid  = t >> 5;
    const int wm   = wid >> 2;               // 0..1 (co)
    const int wn   = wid & 3;                // 0..3 (pos)

    // parity fastest in dispatch -> the 4 parity blocks of a tile co-run and
    // their stride-2 output writes merge in L2 before eviction.
    const int p        = blockIdx.x & 3;     // parity 0..3
    const int co_base  = (blockIdx.x >> 2) * 64;
    const int pos_base = blockIdx.y * 256;   // gridDim.y = 256
    const int pa = p >> 1, pb = p & 1;

    const unsigned short* qwp = qw + (size_t)p * (C_OUT * KF);

    // ---- B gather: one thread per pos row; 4 tap offsets hoisted ----
    const int brow = t;                      // 0..255
    const int bpos = pos_base + brow;
    const int bb   = bpos >> 12;             // batch
    const int bsp  = bpos & (OSPAT - 1);
    const int bi   = bsp >> 6;               // orig row
    const int bj   = bsp & 63;               // orig col
    const float* xb = x + (size_t)bb * (C_IN * 64 * 64);

    const int ih0 = bi + pa - 1, ih1 = ih0 + 1;
    const int iw0 = bj + pb - 1, iw1 = iw0 + 1;
    const bool vr0 = (unsigned)ih0 < 64u, vr1 = (unsigned)ih1 < 64u;
    const bool vc0 = (unsigned)iw0 < 64u, vc1 = (unsigned)iw1 < 64u;
    const bool v00 = vr0 && vc0, v01 = vr0 && vc1, v10 = vr1 && vc0, v11 = vr1 && vc1;
    const int  o00 = (ih0 << 6) + iw0, o01 = (ih0 << 6) + iw1;
    const int  o10 = (ih1 << 6) + iw0, o11 = (ih1 << 6) + iw1;

    // ---- A copy: 4 threads per co row, one async 16B memory->LDS each ----
    const int arow = t >> 2;                 // 0..63
    const int acol = (t & 3) * 8;
    const unsigned short* aga = qwp + (size_t)(co_base + arow) * KF + acol;
    // wave-relative LDS byte offset of this thread's A-tile slot (flat LDS
    // addresses truncate to the LDS offset per ISA 10.2 aperture rules)
    const unsigned lds_dstA = (unsigned)(size_t)&lA[arow][acol];

    const v8f vzero = {0.f, 0.f, 0.f, 0.f, 0.f, 0.f, 0.f, 0.f};
    v8f acc[2][4];
    #pragma unroll
    for (int i = 0; i < 2; ++i)
        #pragma unroll
        for (int j = 0; j < 4; ++j) acc[i][j] = vzero;

    const int khalf = lane >> 4;
    const int l16   = lane & 15;

    for (int k0 = 0; k0 < KF; k0 += 32) {
        // ---- stage A tile (64 co x 32 k) via async global->LDS b128 ----
        {
            const unsigned long long gsrc = (unsigned long long)(size_t)(aga + k0);
            asm volatile("global_load_async_to_lds_b128 %0, %1, off"
                         :: "v"(lds_dstA), "v"(gsrc) : "memory");
        }

        // ---- stage B tile (256 pos x 32 k): 8 ci, 4 taps, packed writes ----
        const int ci0 = k0 >> 2;
        // speculative prefetch of the next ci-block's gather line
        __builtin_prefetch(xb + ((size_t)(ci0 + 8) << 12) + o00, 0, 1);
        #pragma unroll
        for (int c8 = 0; c8 < 8; ++c8) {
            const float* xc = xb + ((size_t)(ci0 + c8) << 12);
            float f00 = v00 ? xc[o00] : 0.0f;
            float f01 = v01 ? xc[o01] : 0.0f;
            float f10 = v10 ? xc[o10] : 0.0f;
            float f11 = v11 ? xc[o11] : 0.0f;
            unsigned u0 = f2bf(f00) | (f2bf(f01) << 16);   // k = ..+0, ..+1
            unsigned u1 = f2bf(f10) | (f2bf(f11) << 16);   // k = ..+2, ..+3
            *(unsigned*)&lB[brow][c8 * 4]     = u0;
            *(unsigned*)&lB[brow][c8 * 4 + 2] = u1;
        }

        // wait for this wave's async LDS writes, then block-wide barrier
        asm volatile("s_wait_asynccnt 0x0" ::: "memory");
        __syncthreads();

        // ---- fragments: two 16B ds reads each (ISA bf16 A/B lane layout) --
        v16bf afr[2], bfr[4];
        #pragma unroll
        for (int fm = 0; fm < 2; ++fm) {
            const v8bf* pa_ = (const v8bf*)&lA[wm * 32 + fm * 16 + l16][0];
            afr[fm] = __builtin_shufflevector(pa_[khalf], pa_[khalf + 2],
                        0,1,2,3,4,5,6,7,8,9,10,11,12,13,14,15);
        }
        #pragma unroll
        for (int fn = 0; fn < 4; ++fn) {
            const v8bf* pb_ = (const v8bf*)&lB[wn * 64 + fn * 16 + l16][0];
            bfr[fn] = __builtin_shufflevector(pb_[khalf], pb_[khalf + 2],
                        0,1,2,3,4,5,6,7,8,9,10,11,12,13,14,15);
        }

        // ---- 8 WMMAs: 32x64 wave tile ----
        #pragma unroll
        for (int fm = 0; fm < 2; ++fm)
            #pragma unroll
            for (int fn = 0; fn < 4; ++fn)
                acc[fm][fn] = __builtin_amdgcn_wmma_f32_16x16x32_bf16(
                    false, afr[fm], false, bfr[fn],
                    (short)0, acc[fm][fn], false, false);
        __syncthreads();
    }

    // ---- epilogue: scatter to upsampled NCHW with parity offsets ----
    #pragma unroll
    for (int fn = 0; fn < 4; ++fn) {
        const int nn   = pos_base + wn * 64 + fn * 16 + l16;
        const int bimg = nn >> 12;
        const int sp   = nn & (OSPAT - 1);
        const int oh   = ((sp >> 6) << 1) + pa;        // 2*i + a
        const int ow   = ((sp & 63) << 1) + pb;        // 2*j + b
        const size_t obase = (((size_t)bimg * C_OUT) << 14) + (size_t)(oh << 7) + ow;
        #pragma unroll
        for (int fm = 0; fm < 2; ++fm) {
            #pragma unroll
            for (int r = 0; r < 8; ++r) {
                const int co = co_base + wm * 32 + fm * 16 + r + khalf * 8;
                out[obase + ((size_t)co << 14)] = acc[fm][fn][r] + bias[co];
            }
        }
    }
}

// ---------------------------------------------------------------------------
// Launcher
// ---------------------------------------------------------------------------
extern "C" void kernel_launch(void* const* d_in, const int* in_sizes, int n_in,
                              void* d_out, int out_size, void* d_ws, size_t ws_size,
                              hipStream_t stream) {
    const float* x      = (const float*)d_in[0];   // 16*256*64*64
    const float* weight = (const float*)d_in[3];   // 589824
    const float* bias   = (const float*)d_in[4];   // 256
    float* out          = (float*)d_out;

    float*          d_acc = (float*)d_ws;                        // 4B
    unsigned short* d_qw  = (unsigned short*)((char*)d_ws + 16); // bf16[4][256][1024] = 2MB

    hipMemsetAsync(d_ws, 0, 16, stream);

    absum_kernel<<<512, 256, 0, stream>>>(weight, d_acc, NW);
    fold_ternarize_kernel<<<(4 * C_OUT * KF) / 256, 256, 0, stream>>>(weight, d_acc, d_qw);

    dim3 grid(16, NPOSF / 256);   // x: parity(4) * coTile(4) fastest; y: 256 pos tiles
    conv_wmma_kernel<<<grid, 256, 0, stream>>>(x, d_qw, bias, out);
}
